// GATLayer_62294205662093
// MI455X (gfx1250) — matrix operand order repaired and checked by reading, not compiled, not run on hardware
//
#include <hip/hip_runtime.h>
#include <hip/hip_bf16.h>

// ---------------- problem constants ----------------
#define N_NODES 10000
#define F_IN    512
#define F_OUT   64
#define ALPHA   0.2f
#define JPAD    10240          // covers 63 tiles of 160 columns (10080)
#define JTILE   160            // columns per attention tile (5 waves x 32)
#define NJB     63             // ceil(10000/160)
#define HT_ROWS 80             // 64 feature rows + 16 rows for the ones-column trick

typedef __attribute__((ext_vector_type(16))) __bf16 v16bf;
typedef __attribute__((ext_vector_type(8)))  __bf16 v8bf;
typedef __attribute__((ext_vector_type(8)))  float  v8f;

// ---------------- workspace layout (bytes), all 32B-aligned ----------------
static constexpr size_t OFF_XBF = 0;                                      // 10000*512 bf16 = 10,240,000
static constexpr size_t OFF_WPK = OFF_XBF + (size_t)N_NODES * F_IN * 2;   // 32768 bf16
static constexpr size_t OFF_H   = OFF_WPK + (size_t)32768 * 2;            // 10000*64 f32
static constexpr size_t OFF_HT  = OFF_H   + (size_t)N_NODES * F_OUT * 4;  // 80*10240 bf16 = 1,638,400
static constexpr size_t OFF_S1  = OFF_HT  + (size_t)HT_ROWS * JPAD * 2;   // 10000 f32
static constexpr size_t OFF_S2  = OFF_S1  + (size_t)N_NODES * 4;          // total ~14.6 MB

// ---------------- kernel 1: X (f32) -> Xbf (bf16) ----------------
__global__ void k_cvt_x(const float* __restrict__ X, __bf16* __restrict__ Xbf) {
  size_t idx = (size_t)blockIdx.x * 256 + threadIdx.x;  // grid sized exactly
  Xbf[idx] = (__bf16)X[idx];
}

// ---------------- kernel 2: pack W into WMMA B-fragment order ----------------
// B (32x16 bf16) order used here: lane = n%16 + 16*(k>=16), element e = k%16 (K contiguous per lane).
__global__ void k_pack_w(const float* __restrict__ W, __bf16* __restrict__ Wpk) {
  int t    = blockIdx.x * 256 + threadIdx.x;   // 0..32767
  int e    = t & 15;
  int lane = (t >> 4) & 31;
  int ct   = (t >> 9) & 3;
  int ks   = t >> 11;                          // 0..15
  int k    = ks * 32 + ((lane >= 16) ? 16 : 0) + e;
  int n    = ct * 16 + (lane & 15);
  Wpk[t]   = (__bf16)W[(size_t)k * F_OUT + n];
}

// ---------------- kernel 3: fill hT padding + the ones/zeros rows 64..79 ----------------
// Elements: 64 rows x 240 pad  (=15360)  then 16 full rows (=163840); total 179200 = 700*256.
__global__ void k_fill_hT(__bf16* __restrict__ hT) {
  int t = blockIdx.x * 256 + threadIdx.x;
  if (t < F_OUT * (JPAD - N_NODES)) {
    int c = t / (JPAD - N_NODES);
    int j = N_NODES + (t % (JPAD - N_NODES));
    hT[(size_t)c * JPAD + j] = (__bf16)0.0f;
  } else {
    int t2 = t - F_OUT * (JPAD - N_NODES);
    int c  = F_OUT + t2 / JPAD;               // 64..79
    int j  = t2 % JPAD;
    float v = (c == F_OUT && j < N_NODES) ? 1.0f : 0.0f;   // row 64 = ones -> Z = P@1
    hT[(size_t)c * JPAD + j] = (__bf16)v;
  }
}

// ---------------- kernel 4: h = X*W via v_wmma_f32_16x16x32_bf16 ----------------
__global__ void __launch_bounds__(128)
k_gemm_h(const __bf16* __restrict__ Xbf, const __bf16* __restrict__ Wpk,
         float* __restrict__ h, __bf16* __restrict__ hT) {
  const int tid  = threadIdx.x;
  const int lane = tid & 31;
  const int wv   = tid >> 5;                 // column tile 0..3
  const int rb   = blockIdx.x;               // row block 0..624
  const int mrow = rb * 16 + (lane & 15);
  const int c0   = (lane < 16) ? 0 : 16;     // A order used here: K contiguous [c0,c0+15]

  v8f acc = {};
#pragma unroll 4
  for (int ks = 0; ks < F_IN / 32; ++ks) {
    const __bf16* ap = Xbf + (size_t)mrow * F_IN + ks * 32 + c0;
    v8bf a_lo = *(const v8bf*)(ap);
    v8bf a_hi = *(const v8bf*)(ap + 8);
    v16bf a = __builtin_shufflevector(a_lo, a_hi, 0,1,2,3,4,5,6,7,8,9,10,11,12,13,14,15);
    v16bf b = *(const v16bf*)(Wpk + ((size_t)(ks * 4 + wv) * 32 + lane) * 16);
    acc = __builtin_amdgcn_wmma_f32_16x16x32_bf16(false, a, false, b, (short)0, acc,
                                                  false, false);
  }
  const int col = wv * 16 + (lane & 15);
  const int hi  = (lane < 16) ? 0 : 8;
  v8bf cb;
#pragma unroll
  for (int r = 0; r < 8; ++r) {
    int row = rb * 16 + hi + r;
    h[(size_t)row * F_OUT + col] = acc[r];
    cb[r] = (__bf16)acc[r];
  }
  *(v8bf*)(hT + (size_t)col * JPAD + rb * 16 + hi) = cb;   // contiguous 16B transposed store
}

// ---------------- kernel 5: s1 = h@a1, s2 = h@a2 ----------------
__global__ void k_s12(const float* __restrict__ h, const float* __restrict__ a,
                      float* __restrict__ s1, float* __restrict__ s2) {
  int i = blockIdx.x * 256 + threadIdx.x;
  if (i >= N_NODES) return;
  float x1 = 0.f, x2 = 0.f;
#pragma unroll 8
  for (int f = 0; f < F_OUT; ++f) {
    float hv = h[(size_t)i * F_OUT + f];
    x1 += hv * a[f];
    x2 += hv * a[F_OUT + f];
  }
  s1[i] = x1;
  s2[i] = x2;
}

// ---------------- kernel 6: single-pass softmax-attention GEMM + ELU ----------------
// 625 blocks x 160 threads (5 waves). Waves 0-3: output col tiles. Wave 4: B = hT rows
// 64..79 = [ones;zeros] so its C column 0 accumulates Z = sum_j P[i,j]. Unnormalized
// P = adj>0 ? exp(lrelu(s1+s2)) : 0 (softmax shift-invariance; exp stays in range here).
__global__ void __launch_bounds__(160)
k_attn(const int* __restrict__ adj, const float* __restrict__ s1,
       const float* __restrict__ s2, const __bf16* __restrict__ hT,
       float* __restrict__ out) {
  __shared__ __bf16 Pt[16][JTILE + 8];   // stride 336B = 84 dwords -> conflict-free b128 reads
  __shared__ float s1_s[16];
  __shared__ float zs[16];

  const int tid  = threadIdx.x;
  const int lane = tid & 31;
  const int wv   = tid >> 5;             // 0..4
  const int i0   = blockIdx.x * 16;

  if (tid < 16) s1_s[tid] = s1[i0 + tid];
  __syncthreads();

  const int r16 = lane & 15;
  const int c0  = (lane < 16) ? 0 : 16;  // A-fragment chunk base
  const int bk0 = (lane < 16) ? 0 : 16;  // B-fragment K base
  const int col = wv * 16 + r16;         // 0..79 (wave 4 hits the ones/zeros rows)
  const int hi  = (lane < 16) ? 0 : 8;

  // ---- prologue: load adj tile 0 into registers (double buffer) ----
  int av[16];
  {
    const int j = tid;
#pragma unroll
    for (int r = 0; r < 16; ++r)
      av[r] = (j < N_NODES) ? adj[(size_t)(i0 + r) * N_NODES + j] : 0;
  }

  v8f acc = {};
  for (int jb = 0; jb < NJB; ++jb) {
    const int j0   = jb * JTILE;
    const int jcur = j0 + tid;
    const float s2j = (jcur < N_NODES) ? s2[jcur] : 0.0f;

    // ---- stage P tile from the prefetched adj registers ----
#pragma unroll 4
    for (int r = 0; r < 16; ++r) {
      float w = 0.0f;
      if (jcur < N_NODES && av[r] > 0) {
        float e = s1_s[r] + s2j;
        e = (e >= 0.0f) ? e : ALPHA * e;
        w = __expf(e);
      }
      Pt[r][tid] = (__bf16)w;
    }

    // ---- issue next tile's adj loads (overlap with WMMA below) ----
    {
      const int jn = j0 + JTILE + tid;
#pragma unroll
      for (int r = 0; r < 16; ++r)
        av[r] = (jn < N_NODES) ? adj[(size_t)(i0 + r) * N_NODES + jn] : 0;
    }
    // L2 prefetch two tiles ahead (tid&15 row assignment covers all 16 rows/lines)
    __builtin_prefetch(adj + (size_t)(i0 + (tid & 15)) * N_NODES + j0 + 2 * JTILE + tid, 0, 1);

    __syncthreads();
    // ---- 5 WMMA k-steps over the staged 16x160 tile ----
#pragma unroll
    for (int ks = 0; ks < 5; ++ks) {
      const __bf16* ap = &Pt[r16][ks * 32 + c0];
      v8bf a_lo = *(const v8bf*)(ap);
      v8bf a_hi = *(const v8bf*)(ap + 8);
      v16bf a = __builtin_shufflevector(a_lo, a_hi, 0,1,2,3,4,5,6,7,8,9,10,11,12,13,14,15);
      v16bf b = *(const v16bf*)(hT + (size_t)col * JPAD + j0 + ks * 32 + bk0);
      acc = __builtin_amdgcn_wmma_f32_16x16x32_bf16(false, a, false, b, (short)0, acc,
                                                    false, false);
    }
    __syncthreads();
  }

  // ---- wave 4 publishes Z (its C column 0 = row sums), then all normalize ----
  if (wv == 4 && r16 == 0) {
#pragma unroll
    for (int r = 0; r < 8; ++r) zs[hi + r] = acc[r];
  }
  __syncthreads();

  if (wv < 4) {
#pragma unroll
    for (int r = 0; r < 8; ++r) {
      int   i = i0 + hi + r;
      float z = zs[hi + r];
      float v = (z > 0.0f) ? acc[r] / z : 0.0f;
      v = (v > 0.0f) ? v : (__expf(v) - 1.0f);
      out[(size_t)i * F_OUT + col] = v;
    }
  }
}

// ---------------- launcher ----------------
extern "C" void kernel_launch(void* const* d_in, const int* in_sizes, int n_in,
                              void* d_out, int out_size, void* d_ws, size_t ws_size,
                              hipStream_t stream) {
  const float* X   = (const float*)d_in[0];   // (10000, 512)
  const float* W   = (const float*)d_in[1];   // (512, 64)
  const float* a   = (const float*)d_in[2];   // (128, 1)
  const int*   adj = (const int*)d_in[3];     // (10000, 10000)
  float* out = (float*)d_out;                 // (10000, 64)

  char* ws = (char*)d_ws;
  __bf16* Xbf = (__bf16*)(ws + OFF_XBF);
  __bf16* Wpk = (__bf16*)(ws + OFF_WPK);
  float*  h   = (float*) (ws + OFF_H);
  __bf16* hT  = (__bf16*)(ws + OFF_HT);
  float*  s1  = (float*) (ws + OFF_S1);
  float*  s2  = (float*) (ws + OFF_S2);

  k_cvt_x  <<<(N_NODES * F_IN) / 256, 256, 0, stream>>>(X, Xbf);
  k_pack_w <<<32768 / 256,            256, 0, stream>>>(W, Wpk);
  k_fill_hT<<<(F_OUT * (JPAD - N_NODES) + 16 * JPAD) / 256, 256, 0, stream>>>(hT);
  k_gemm_h <<<N_NODES / 16, 128, 0, stream>>>(Xbf, Wpk, h, hT);
  k_s12    <<<(N_NODES + 255) / 256, 256, 0, stream>>>(h, a, s1, s2);
  k_attn   <<<N_NODES / 16, 160, 0, stream>>>(adj, s1, s2, hT, out);
}
